// DAG_RNN_se_12232066859244
// MI455X (gfx1250) — compile-verified, matrix operand order checked by reading, not run on hardware
//
#include <hip/hip_runtime.h>

#define Bdim 32
#define Cdim 512
#define Hdim 32
#define Wdim 32

typedef __attribute__((ext_vector_type(16))) _Float16 v16h;
typedef __attribute__((ext_vector_type(8)))  _Float16 v8h;
typedef __attribute__((ext_vector_type(8)))  float    v8f;

// Load a 16x32 (f16) A/B fragment for v_wmma_f32_16x16x32_f16 from a row-major
// [rows x Cdim] f16 matrix. ISA layout (CDNA5 ISA):
//   lanes 0-15 : row r0+lane,    K = kk+{0..7}  (v0-3) and kk+{16..23} (v4-7)
//   lanes 16-31: row r0+lane-16, K = kk+{8..15} (v0-3) and kk+{24..31} (v4-7)
// => two contiguous 16-byte loads per lane (ds_load_b128 / global_load_b128).
__device__ __forceinline__ v16h load_frag(const _Float16* p, int r0, int kk, int lane) {
  const int sel = (lane >> 4) & 1;
  const int row = r0 + (lane & 15);
  const _Float16* base = p + row * Cdim + kk + sel * 8;
  v8h lo = *(const v8h*)(base);
  v8h hi = *(const v8h*)(base + 16);
  v16h out;
#pragma unroll
  for (int t = 0; t < 8; ++t) { out[t] = lo[t]; out[t + 8] = hi[t]; }
  return out;
}

// One-time: transpose + convert W_hh and W_yh to f16, [N,K] row-major so that
// B-matrix fragment loads are per-lane contiguous along K.
__global__ void convert_weights_kernel(const float* __restrict__ whh,
                                       const float* __restrict__ wyh,
                                       _Float16* __restrict__ whhT,
                                       _Float16* __restrict__ wyhT) {
  int idx = blockIdx.x * blockDim.x + threadIdx.x;
  if (idx < Cdim * Cdim) {
    int n = idx / Cdim;   // out-channel (row of W^T)
    int k = idx % Cdim;   // in-channel  (col of W^T)
    whhT[idx] = (_Float16)whh[k * Cdim + n];
    wyhT[idx] = (_Float16)wyh[k * Cdim + n];
  }
}

// One workgroup (8 wave32s) per DAG cell on the current anti-diagonal d.
// Phase 1: hsum = h_up + h_left -> LDS (v_pk_add_f16)
// Phase 2: h = relu(x + hsum @ Whh)  (WMMA f16->f32), h -> LDS + global f16
// Phase 3: y = h @ Wyh + bias        (WMMA f16->f32), scatter to [B,C,H,W]
// Each wave: 1 M-tile x 4 N-tiles per step; A fragment loaded once per K-step
// and reused across 4 back-to-back WMMAs on 4 accumulators.
__global__ __launch_bounds__(256) void dag_cell_kernel(
    const float*    __restrict__ x,
    const _Float16* __restrict__ whhT,
    const _Float16* __restrict__ wyhT,
    const float*    __restrict__ bias,
    _Float16*       __restrict__ hidden,
    float*          __restrict__ out,
    int d, int i0)
{
  __shared__ _Float16 hsum[Bdim * Cdim];  // 32 KB
  __shared__ _Float16 hcur[Bdim * Cdim];  // 32 KB

  const int i    = i0 + (int)blockIdx.x;
  const int j    = d - i;
  const int tid  = threadIdx.x;
  const int lane = tid & 31;
  const int wave = tid >> 5;

  const bool has_up   = (i > 0);
  const bool has_left = (j > 0);
  const _Float16* h_up   = hidden + (size_t)(has_up   ? (((i - 1) * Wdim + j) * (Bdim * Cdim)) : 0);
  const _Float16* h_left = hidden + (size_t)(has_left ? ((i * Wdim + (j - 1)) * (Bdim * Cdim)) : 0);
  _Float16* h_out = hidden + (size_t)((i * Wdim + j) * (Bdim * Cdim));

  // ---- Phase 1: sum of predecessor hidden states into LDS (packed f16) ----
  {
    const v8h* upv = (const v8h*)h_up;
    const v8h* lfv = (const v8h*)h_left;
    v8h* dstv = (v8h*)hsum;
    const v8h zero = {};
#pragma unroll
    for (int c = 0; c < (Bdim * Cdim / 8) / 256; ++c) {
      int idx = tid + c * 256;
      v8h u = has_up   ? upv[idx] : zero;
      v8h l = has_left ? lfv[idx] : zero;
      dstv[idx] = u + l;                       // v_pk_add_f16 x4
    }
  }
  __syncthreads();

  const int m0    = (wave & 1) * 16;           // M-tile (batch rows)
  const int wc    = wave >> 1;                 // N-tile-group column 0..3
  const int sel   = (lane >> 4) & 1;
  const int col16 = lane & 15;
  const int rbase = m0 + sel * 8;              // D-fragment row base for this lane
  const int xoff  = i * Wdim + j;

  // ---- Phase 2: recurrence GEMM + x + ReLU ----
#pragma unroll 1
  for (int t = 0; t < 2; ++t) {
    const int nbase = (wc + t * 4) * 64;       // 4 consecutive 16-wide N-tiles
    v8f acc[4] = {};
#pragma unroll 2
    for (int kk = 0; kk < Cdim; kk += 32) {
      v16h a = load_frag(hsum, m0, kk, lane);              // ds_load_b128 x2
      __builtin_prefetch(whhT + (nbase + col16) * Cdim + kk + 64, 0, 3);
#pragma unroll
      for (int q = 0; q < 4; ++q) {
        v16h b = load_frag(whhT, nbase + q * 16, kk, lane); // global_load_b128 x2 (L2-hot)
        acc[q] = __builtin_amdgcn_wmma_f32_16x16x32_f16(
            false, a, false, b, (short)0, acc[q], false, false);
      }
    }
#pragma unroll
    for (int q = 0; q < 4; ++q) {
      const int col = nbase + q * 16 + col16;
#pragma unroll
      for (int r = 0; r < 8; ++r) {
        const int row = rbase + r;                          // batch index
        float xv = x[(size_t)row * (Cdim * Hdim * Wdim) + (size_t)col * (Hdim * Wdim) + xoff];
        float hv = acc[q][r] + xv;
        hv = hv > 0.f ? hv : 0.f;
        _Float16 hq = (_Float16)hv;
        hcur[row * Cdim + col]  = hq;                       // ds_store_b16
        h_out[row * Cdim + col] = hq;                       // global f16 for successors
      }
    }
  }
  __syncthreads();

  // ---- Phase 3: fused output GEMM y = h @ Wyh + bias ----
#pragma unroll 1
  for (int t = 0; t < 2; ++t) {
    const int nbase = (wc + t * 4) * 64;
    v8f acc[4] = {};
#pragma unroll 2
    for (int kk = 0; kk < Cdim; kk += 32) {
      v16h a = load_frag(hcur, m0, kk, lane);
      __builtin_prefetch(wyhT + (nbase + col16) * Cdim + kk + 64, 0, 3);
#pragma unroll
      for (int q = 0; q < 4; ++q) {
        v16h b = load_frag(wyhT, nbase + q * 16, kk, lane);
        acc[q] = __builtin_amdgcn_wmma_f32_16x16x32_f16(
            false, a, false, b, (short)0, acc[q], false, false);
      }
    }
#pragma unroll
    for (int q = 0; q < 4; ++q) {
      const int col = nbase + q * 16 + col16;
      const float bv = bias[col];
#pragma unroll
      for (int r = 0; r < 8; ++r) {
        const int row = rbase + r;
        out[(size_t)(row * Cdim + col) * (Hdim * Wdim) + xoff] = acc[q][r] + bv;
      }
    }
  }
}

extern "C" void kernel_launch(void* const* d_in, const int* in_sizes, int n_in,
                              void* d_out, int out_size, void* d_ws, size_t ws_size,
                              hipStream_t stream) {
  const float* x    = (const float*)d_in[0];
  const float* whh  = (const float*)d_in[1];
  const float* wyh  = (const float*)d_in[2];
  const float* bias = (const float*)d_in[3];
  float* out = (float*)d_out;

  // Workspace layout: WhhT f16 (512 KB) | WyhT f16 (512 KB) | hidden f16 (32 MB)
  char* ws = (char*)d_ws;
  _Float16* whhT   = (_Float16*)ws;
  _Float16* wyhT   = (_Float16*)(ws + (size_t)Cdim * Cdim * sizeof(_Float16));
  _Float16* hidden = (_Float16*)(ws + (size_t)2 * Cdim * Cdim * sizeof(_Float16));

  convert_weights_kernel<<<(Cdim * Cdim + 255) / 256, 256, 0, stream>>>(whh, wyh, whhT, wyhT);

  // Wavefront sweep: one launch per anti-diagonal (graph replay amortizes this).
  for (int d = 0; d <= Hdim + Wdim - 2; ++d) {
    int ilo = d - (Wdim - 1); if (ilo < 0) ilo = 0;
    int ihi = (d < Hdim - 1) ? d : (Hdim - 1);
    int ncells = ihi - ilo + 1;
    dag_cell_kernel<<<ncells, 256, 0, stream>>>(x, whhT, wyhT, bias, hidden, out, d, ilo);
  }
}
